// InfluenceEncoder_21534966022835
// MI455X (gfx1250) — matrix operand order repaired and checked by reading, not compiled, not run on hardware
//
#include <hip/hip_runtime.h>

// CDNA5 / gfx1250: wave32, WMMA 16x16x32 f16 -> f32 accumulate.
typedef __attribute__((ext_vector_type(16))) _Float16 v16h;
typedef __attribute__((ext_vector_type(8)))  _Float16 v8h;
typedef __attribute__((ext_vector_type(8)))  float    v8f;

#define BATCH  4096
#define NROW   257
#define NAGENT 256
#define DIM    16
#define AROW   32     // padded A row in LDS: 16 data + 1.0 bias col + zeros
#define HID    128

__launch_bounds__(256)
__global__ void influence_encoder_kernel(
    const float* __restrict__ x,      // (B, 257, 16)
    const float* __restrict__ W_fc,   // (128, 16)
    const float* __restrict__ b_fc,   // (128)
    const float* __restrict__ W_agg,  // (128, 128)
    const float* __restrict__ b_agg,  // (128)
    float* __restrict__ out)          // (B, 128)
{
  __shared__ alignas(16) _Float16 s_a[NAGENT * AROW]; // agents f16, K padded
  __shared__ alignas(16) float s_w[NAGENT];           // unnormalized weights
  __shared__ alignas(16) float s_agg[HID];            // weighted emb reduction
  __shared__ float s_wsum;

  const int tid  = threadIdx.x;
  const int lane = tid & 31;
  const int wave = tid >> 5;          // 0..7, one H column tile per wave
  const int b    = blockIdx.x;

  const float* xb = x + (size_t)b * NROW * DIM;

  if (tid < HID) s_agg[tid] = 0.0f;
  if (tid == 0)  s_wsum = 0.0f;

  // Warm L2/L0 for the stage-2 weight row this thread will need.
  if (tid < HID) __builtin_prefetch(W_agg + (size_t)tid * HID, 0, 3);

  // ---- load one agent row per thread; weight + f32->f16 LDS staging ----
  const float ego0 = xb[0];
  const float ego1 = xb[1];
  const float4* rp = reinterpret_cast<const float4*>(xb + (size_t)(tid + 1) * DIM);
  float4 r0 = rp[0], r1 = rp[1], r2 = rp[2], r3 = rp[3];

  float dx = r0.x - ego0, dy = r0.y - ego1;
  float dist = sqrtf(dx * dx + dy * dy);
  float wgt  = 1.0f / (dist + 1.0f);
  if (r1.z == 1.0f) wgt *= 5.0f;      // agents[:, 6] == 1 -> 5x boost
  s_w[tid] = wgt;

  float t16[16] = { r0.x, r0.y, r0.z, r0.w, r1.x, r1.y, r1.z, r1.w,
                    r2.x, r2.y, r2.z, r2.w, r3.x, r3.y, r3.z, r3.w };
  v8h lo, hi;
  #pragma unroll
  for (int i = 0; i < 8; ++i) {
    lo[i] = (_Float16)t16[i];
    hi[i] = (_Float16)t16[i + 8];
  }
  v8h one0 = {};
  one0[0] = (_Float16)1.0f;           // K=16 bias column of A
  const v8h zero8 = {};
  v8h* srow = reinterpret_cast<v8h*>(&s_a[tid * AROW]);
  srow[0] = lo;                        // K 0..7
  srow[1] = hi;                        // K 8..15
  srow[2] = one0;                      // K 16..23 (16 -> 1.0)
  srow[3] = zero8;                     // K 24..31

  // wave32 reduction of the weight sum, one LDS atomic per wave
  float ws = wgt;
  #pragma unroll
  for (int off = 16; off > 0; off >>= 1) ws += __shfl_xor(ws, off, 32);
  if (lane == 0) atomicAdd(&s_wsum, ws);

  __syncthreads();

  // ---- stage 1: emb = relu(agents @ W_fc^T + b_fc); agg += w * emb (WMMA) ----
  const int nlocal = lane & 15;        // N column within tile (and M row for A)
  const int grp    = lane >> 4;        // lane half
  const int hcol   = wave * 16 + nlocal;

  // B fragment (32x16 f16): lanes 0-15 carry K=0..15 = W_fc[hcol][0..15]
  // (VGPR v holds K=2v,2v+1 in lo/hi); lanes 16-31 carry K=16..31, where
  // B[16][n] = b_fc[hcol] pairs with the constant-1 column of A (fused bias).
  v16h bfrag = {};
  if (lane < 16) {
    const float4* wp = reinterpret_cast<const float4*>(W_fc + (size_t)hcol * DIM);
    float4 q0 = wp[0], q1 = wp[1], q2 = wp[2], q3 = wp[3];
    float tw[16] = { q0.x, q0.y, q0.z, q0.w, q1.x, q1.y, q1.z, q1.w,
                     q2.x, q2.y, q2.z, q2.w, q3.x, q3.y, q3.z, q3.w };
    #pragma unroll
    for (int i = 0; i < 16; ++i) bfrag[i] = (_Float16)tw[i];
  } else {
    bfrag[0] = (_Float16)b_fc[hcol];   // K=16 row of B
  }

  union AFrag { v16h v; v8h h[2]; };

  float a0 = 0.0f, a1 = 0.0f, a2 = 0.0f, a3 = 0.0f;
  #pragma unroll 4
  for (int r = 0; r < 16; ++r) {
    // A fragment (16x32 f16), row M=nlocal of tile r. Lane-half grp selects
    // K0..15 / K16..31 (h[0] = first 8, h[1] = next 8): two ds_load_b128
    // straight into the contiguous operand block.
    const int abase = (16 * r + nlocal) * AROW + grp * 16;
    AFrag af;
    af.h[0] = *reinterpret_cast<const v8h*>(&s_a[abase]);
    af.h[1] = *reinterpret_cast<const v8h*>(&s_a[abase + 8]);

    v8f c = {};
    c = __builtin_amdgcn_wmma_f32_16x16x32_f16(
        false, af.v, false, bfrag, (short)0, c, false, false);

    // C/D layout: element e, lane L -> M = e + 8*grp, N = L&15.
    const float4* wp4 = reinterpret_cast<const float4*>(&s_w[16 * r + grp * 8]);
    float4 w0 = wp4[0], w1 = wp4[1];

    a0 = fmaf(w0.x, fmaxf(c[0], 0.0f), a0);
    a1 = fmaf(w0.y, fmaxf(c[1], 0.0f), a1);
    a2 = fmaf(w0.z, fmaxf(c[2], 0.0f), a2);
    a3 = fmaf(w0.w, fmaxf(c[3], 0.0f), a3);
    a0 = fmaf(w1.x, fmaxf(c[4], 0.0f), a0);
    a1 = fmaf(w1.y, fmaxf(c[5], 0.0f), a1);
    a2 = fmaf(w1.z, fmaxf(c[6], 0.0f), a2);
    a3 = fmaf(w1.w, fmaxf(c[7], 0.0f), a3);
  }
  atomicAdd(&s_agg[hcol], (a0 + a1) + (a2 + a3));

  __syncthreads();

  // ---- stage 2: out = relu(agg / wsum) @ W_agg^T + b_agg (L2-resident) ----
  // float4 row reads (global_load_b128) + 4 independent accumulators.
  if (tid < HID) {
    const float winv = 1.0f / (s_wsum + 1e-10f);
    const float4* wrow = reinterpret_cast<const float4*>(W_agg + (size_t)tid * HID);
    const float4* ragg = reinterpret_cast<const float4*>(s_agg);
    float o0 = b_agg[tid], o1 = 0.0f, o2 = 0.0f, o3 = 0.0f;
    #pragma unroll 8
    for (int h4 = 0; h4 < HID / 4; ++h4) {
      float4 wv = wrow[h4];
      float4 av = ragg[h4];
      o0 = fmaf(fmaxf(av.x * winv, 0.0f), wv.x, o0);
      o1 = fmaf(fmaxf(av.y * winv, 0.0f), wv.y, o1);
      o2 = fmaf(fmaxf(av.z * winv, 0.0f), wv.z, o2);
      o3 = fmaf(fmaxf(av.w * winv, 0.0f), wv.w, o3);
    }
    out[(size_t)b * HID + tid] = (o0 + o1) + (o2 + o3);
  }
}

extern "C" void kernel_launch(void* const* d_in, const int* in_sizes, int n_in,
                              void* d_out, int out_size, void* d_ws, size_t ws_size,
                              hipStream_t stream) {
  (void)in_sizes; (void)n_in; (void)out_size; (void)d_ws; (void)ws_size;
  const float* x     = (const float*)d_in[0];
  const float* W_fc  = (const float*)d_in[1];
  const float* b_fc  = (const float*)d_in[2];
  const float* W_agg = (const float*)d_in[3];
  const float* b_agg = (const float*)d_in[4];
  float* out = (float*)d_out;

  influence_encoder_kernel<<<dim3(BATCH), dim3(256), 0, stream>>>(
      x, W_fc, b_fc, W_agg, b_agg, out);
}
